// Survival_prediction_15101105013143
// MI455X (gfx1250) — compile-verified, matrix operand order checked by reading, not run on hardware
//
#include <hip/hip_runtime.h>
#include <math.h>

typedef __attribute__((ext_vector_type(2)))  float  v2f;
typedef __attribute__((ext_vector_type(8)))  float  v8f;
typedef __attribute__((ext_vector_type(4)))  __bf16 v4bf;
typedef __attribute__((ext_vector_type(8)))  __bf16 v8bf;
typedef __attribute__((ext_vector_type(16))) __bf16 v16bf;

#define NT 65536
#define KDIM 256

#define CAT16(lo, hi) \
    __builtin_shufflevector(lo, hi, 0, 1, 2, 3, 4, 5, 6, 7, 8, 9, 10, 11, 12, 13, 14, 15)

__device__ __forceinline__ v8f wmma_bf16(v16bf a, v16bf b, v8f c) {
    return __builtin_amdgcn_wmma_f32_16x16x32_bf16(
        false, a, false, b, (short)0, c, false, false);
}

// ---------------------------------------------------------------------------
// Per source: split w and beta rows into bf16 hi/lo planes, compute ||w_k||^2.
// One block per k-component (256 blocks).
// ---------------------------------------------------------------------------
__global__ void split_kernel(const float* __restrict__ w,
                             const float* __restrict__ beta,
                             __bf16* __restrict__ wh, __bf16* __restrict__ wl,
                             __bf16* __restrict__ bh, __bf16* __restrict__ bl,
                             float* __restrict__ wn, int P) {
    __shared__ float acc;
    const int k = blockIdx.x;
    if (threadIdx.x == 0) acc = 0.f;
    __syncthreads();
    const float* wrow = w    + (size_t)k * P;
    const float* brow = beta + (size_t)k * P;
    float s = 0.f;
    for (int j = threadIdx.x; j < P; j += blockDim.x) {
        float v  = wrow[j];
        __bf16 h = (__bf16)v;
        wh[(size_t)k * P + j] = h;
        wl[(size_t)k * P + j] = (__bf16)(v - (float)h);
        s += v * v;
        float b   = brow[j];
        __bf16 hb = (__bf16)b;
        bh[(size_t)k * P + j] = hb;
        bl[(size_t)k * P + j] = (__bf16)(b - (float)hb);
    }
    atomicAdd(&acc, s);
    __syncthreads();
    if (threadIdx.x == 0) wn[k] = acc;
}

// ---------------------------------------------------------------------------
// Fused per-source kernel (bf16-split WMMA GEMMs, fp32 accumulate).
// Block: 256 threads = 8 waves on a 32-row tile (2 row-tiles of 16); wave wv
// owns k-columns [wv*32, wv*32+32) as two 16-col tiles.  Per 32-K chunk:
// 8 ds_load_b128 + 16 global_load_b128 + 24 WMMAs on 8 independent
// accumulator chains (B fragments amortized over 2 row-tiles).
// ---------------------------------------------------------------------------
template <int P>
__global__ __launch_bounds__(256)
void source_kernel(const float* __restrict__ x,
                   const float* __restrict__ alpha,
                   const float* __restrict__ sig,
                   const float* __restrict__ eta,
                   const float* __restrict__ gam,
                   const __bf16* __restrict__ wh, const __bf16* __restrict__ wl,
                   const __bf16* __restrict__ bh, const __bf16* __restrict__ bl,
                   const float* __restrict__ wn,
                   float* __restrict__ out, int src) {
    __shared__ __align__(16) __bf16 xh[32 * P];   // x tile, bf16 hi plane
    __shared__ __align__(16) __bf16 xl[32 * P];   // x tile, bf16 lo plane
    __shared__ float xnorm[32];
    __shared__ float hxA[32], muxA[32], s2A[32];

    const int tid  = threadIdx.x;
    const int lane = tid & 31;
    const int wv   = tid >> 5;
    const int row0 = blockIdx.x * 32;

    if (tid < 32) { xnorm[tid] = 0.f; hxA[tid] = 0.f; muxA[tid] = 0.f; s2A[tid] = 0.f; }
    __syncthreads();

    // Load + split the 32 x P tile of x; accumulate ||x_row||^2 on the fly.
    {
        const float4* xg = (const float4*)(x + (size_t)row0 * P);
        const int n4 = 32 * P / 4;
        for (int i = tid; i < n4; i += 256) {
            float4 v  = xg[i];
            int  base = 4 * i;
            int  r    = base / P;               // row is constant within a float4
            atomicAdd(&xnorm[r], v.x * v.x + v.y * v.y + v.z * v.z + v.w * v.w);
            __bf16 h0 = (__bf16)v.x, h1 = (__bf16)v.y, h2 = (__bf16)v.z, h3 = (__bf16)v.w;
            v4bf hv = {h0, h1, h2, h3};
            v4bf lv = {(__bf16)(v.x - (float)h0), (__bf16)(v.y - (float)h1),
                       (__bf16)(v.z - (float)h2), (__bf16)(v.w - (float)h3)};
            *(v4bf*)(xh + base) = hv;
            *(v4bf*)(xl + base) = lv;
        }
    }
    __syncthreads();

    const int half = lane >> 4;          // 0: lanes 0-15, 1: lanes 16-31
    const int l15  = lane & 15;
    const int koff = half * 8;           // 16-bit WMMA fragment K offset

    const int col0 = wv * 32 + l15;      // this lane's k-column, col-tile 0
    const int col1 = col0 + 16;          // col-tile 1

    v8f accW[2][2], accB[2][2];          // [row-tile][col-tile]
    accW[0][0] = (v8f){0.f, 0.f, 0.f, 0.f, 0.f, 0.f, 0.f, 0.f};
    accW[0][1] = accW[0][0]; accW[1][0] = accW[0][0]; accW[1][1] = accW[0][0];
    accB[0][0] = accW[0][0]; accB[0][1] = accW[0][0];
    accB[1][0] = accW[0][0]; accB[1][1] = accW[0][0];

    const __bf16* a0h = xh + l15 * P + koff;          // A frags: rows l15, 16+l15
    const __bf16* a0l = xl + l15 * P + koff;
    const __bf16* a1h = xh + (16 + l15) * P + koff;
    const __bf16* a1l = xl + (16 + l15) * P + koff;
    const __bf16* w0h = wh + (size_t)col0 * P + koff;  // B frags: n = l15
    const __bf16* w0l = wl + (size_t)col0 * P + koff;
    const __bf16* b0h = bh + (size_t)col0 * P + koff;
    const __bf16* b0l = bl + (size_t)col0 * P + koff;
    const __bf16* w1h = wh + (size_t)col1 * P + koff;
    const __bf16* w1l = wl + (size_t)col1 * P + koff;
    const __bf16* b1h = bh + (size_t)col1 * P + koff;
    const __bf16* b1l = bl + (size_t)col1 * P + koff;

#pragma unroll 1
    for (int kk = 0; kk < P; kk += 32) {
        v16bf A0h = CAT16(*(const v8bf*)(a0h + kk), *(const v8bf*)(a0h + kk + 16));
        v16bf A0l = CAT16(*(const v8bf*)(a0l + kk), *(const v8bf*)(a0l + kk + 16));
        v16bf A1h = CAT16(*(const v8bf*)(a1h + kk), *(const v8bf*)(a1h + kk + 16));
        v16bf A1l = CAT16(*(const v8bf*)(a1l + kk), *(const v8bf*)(a1l + kk + 16));
        v16bf W0h = CAT16(*(const v8bf*)(w0h + kk), *(const v8bf*)(w0h + kk + 16));
        v16bf W0l = CAT16(*(const v8bf*)(w0l + kk), *(const v8bf*)(w0l + kk + 16));
        v16bf B0h = CAT16(*(const v8bf*)(b0h + kk), *(const v8bf*)(b0h + kk + 16));
        v16bf B0l = CAT16(*(const v8bf*)(b0l + kk), *(const v8bf*)(b0l + kk + 16));
        v16bf W1h = CAT16(*(const v8bf*)(w1h + kk), *(const v8bf*)(w1h + kk + 16));
        v16bf W1l = CAT16(*(const v8bf*)(w1l + kk), *(const v8bf*)(w1l + kk + 16));
        v16bf B1h = CAT16(*(const v8bf*)(b1h + kk), *(const v8bf*)(b1h + kk + 16));
        v16bf B1l = CAT16(*(const v8bf*)(b1l + kk), *(const v8bf*)(b1l + kk + 16));
        // term hi*hi : 8 distinct chains
        accW[0][0] = wmma_bf16(A0h, W0h, accW[0][0]);
        accW[0][1] = wmma_bf16(A0h, W1h, accW[0][1]);
        accB[0][0] = wmma_bf16(A0h, B0h, accB[0][0]);
        accB[0][1] = wmma_bf16(A0h, B1h, accB[0][1]);
        accW[1][0] = wmma_bf16(A1h, W0h, accW[1][0]);
        accW[1][1] = wmma_bf16(A1h, W1h, accW[1][1]);
        accB[1][0] = wmma_bf16(A1h, B0h, accB[1][0]);
        accB[1][1] = wmma_bf16(A1h, B1h, accB[1][1]);
        // term hi*lo
        accW[0][0] = wmma_bf16(A0h, W0l, accW[0][0]);
        accW[0][1] = wmma_bf16(A0h, W1l, accW[0][1]);
        accB[0][0] = wmma_bf16(A0h, B0l, accB[0][0]);
        accB[0][1] = wmma_bf16(A0h, B1l, accB[0][1]);
        accW[1][0] = wmma_bf16(A1h, W0l, accW[1][0]);
        accW[1][1] = wmma_bf16(A1h, W1l, accW[1][1]);
        accB[1][0] = wmma_bf16(A1h, B0l, accB[1][0]);
        accB[1][1] = wmma_bf16(A1h, B1l, accB[1][1]);
        // term lo*hi  (lo*lo dropped, ~2^-16 rel)
        accW[0][0] = wmma_bf16(A0l, W0h, accW[0][0]);
        accW[0][1] = wmma_bf16(A0l, W1h, accW[0][1]);
        accB[0][0] = wmma_bf16(A0l, B0h, accB[0][0]);
        accB[0][1] = wmma_bf16(A0l, B1h, accB[0][1]);
        accW[1][0] = wmma_bf16(A1l, W0h, accW[1][0]);
        accW[1][1] = wmma_bf16(A1l, W1h, accW[1][1]);
        accB[1][0] = wmma_bf16(A1l, B0h, accB[1][0]);
        accB[1][1] = wmma_bf16(A1l, B1h, accB[1][1]);
    }

    // Elementwise + per-row accumulation (lane = column, rows r + half*8).
    float hx_r[2][8], mux_r[2][8], s2_r[2][8];
#pragma unroll
    for (int rt = 0; rt < 2; ++rt)
#pragma unroll
        for (int r = 0; r < 8; ++r) { hx_r[rt][r] = 0.f; mux_r[rt][r] = 0.f; s2_r[rt][r] = 0.f; }

#pragma unroll
    for (int ct = 0; ct < 2; ++ct) {
        const int col   = ct ? col1 : col0;
        const float al  = alpha[col];
        const float g   = gam[col];
        const float et  = eta[col];
        const float sg  = sig[col];
        const float wnc = wn[col];
        const float g2 = g * g, h = et * et, s2c = sg * sg;
#pragma unroll
        for (int rt = 0; rt < 2; ++rt) {
#pragma unroll
            for (int r = 0; r < 8; ++r) {
                int rw   = rt * 16 + r + half * 8;
                float xw = accW[rt][ct][r];
                float d2 = xnorm[rw] - 2.f * xw + wnc;
                float a  = __expf(-g2 * d2);
                float ah = a * h;
                float mu = accB[rt][ct][r] + al;
                hx_r[rt][r]  += ah;
                mux_r[rt][r] += mu * ah;
                s2_r[rt][r]  += s2c * ah * ah;
            }
        }
    }

    // Reduce across the 16 lanes of each half (columns), then across waves.
#pragma unroll
    for (int rt = 0; rt < 2; ++rt) {
#pragma unroll
        for (int r = 0; r < 8; ++r) {
#pragma unroll
            for (int m = 1; m < 16; m <<= 1) {
                hx_r[rt][r]  += __shfl_xor(hx_r[rt][r],  m, 32);
                mux_r[rt][r] += __shfl_xor(mux_r[rt][r], m, 32);
                s2_r[rt][r]  += __shfl_xor(s2_r[rt][r],  m, 32);
            }
            if (l15 == 0) {
                int rw = rt * 16 + r + half * 8;
                atomicAdd(&hxA[rw],  hx_r[rt][r]);
                atomicAdd(&muxA[rw], mux_r[rt][r]);
                atomicAdd(&s2A[rw],  s2_r[rt][r]);
            }
        }
    }
    __syncthreads();

    if (tid < 32) {
        int grow = row0 + tid;
        float hx = hxA[tid];
        out[(size_t)src * NT + grow]       = muxA[tid] / hx;        // mux_t
        out[(size_t)(4 + src) * NT + grow] = s2A[tid] / (hx * hx);  // sig2x_t
        out[(size_t)(8 + src) * NT + grow] = hx;                    // hx_t
    }
}

// ---------------------------------------------------------------------------
// Combine across sources (rows 3 / 7 / 11 of the flattened output blocks).
// ---------------------------------------------------------------------------
__global__ void combine_kernel(float* __restrict__ out,
                               const float* __restrict__ disc0,
                               const float* __restrict__ disc1,
                               const float* __restrict__ disc2) {
    int t = blockIdx.x * blockDim.x + threadIdx.x;
    float dc[3];
    dc[0] = 1.f / (1.f + __expf(-disc0[0]));
    dc[1] = 1.f / (1.f + __expf(-disc1[0]));
    dc[2] = 1.f / (1.f + __expf(-disc2[0]));
    float denom = 0.f, mc = 0.f, sc = 0.f;
#pragma unroll
    for (int s = 0; s < 3; ++s) {
        float hx   = out[(size_t)(8 + s) * NT + t];
        float mux  = out[(size_t)s * NT + t];
        float sg2  = out[(size_t)(4 + s) * NT + t];
        float hxdc = hx * dc[s];
        denom += hxdc;
        mc    += mux * hxdc;
        sc    += sg2 * hxdc * hxdc;
    }
    out[(size_t)3 * NT + t]  = mc / denom;
    out[(size_t)7 * NT + t]  = sc / (denom * denom);
    out[(size_t)11 * NT + t] = denom;
}

// ---------------------------------------------------------------------------
extern "C" void kernel_launch(void* const* d_in, const int* in_sizes, int n_in,
                              void* d_out, int out_size, void* d_ws, size_t ws_size,
                              hipStream_t stream) {
    (void)in_sizes; (void)n_in; (void)out_size; (void)ws_size;
    float* out = (float*)d_out;

    const float* x[3]     = {(const float*)d_in[0],  (const float*)d_in[8],  (const float*)d_in[16]};
    const float* alpha[3] = {(const float*)d_in[1],  (const float*)d_in[9],  (const float*)d_in[17]};
    const float* beta[3]  = {(const float*)d_in[2],  (const float*)d_in[10], (const float*)d_in[18]};
    const float* sig[3]   = {(const float*)d_in[3],  (const float*)d_in[11], (const float*)d_in[19]};
    const float* eta[3]   = {(const float*)d_in[4],  (const float*)d_in[12], (const float*)d_in[20]};
    const float* gam[3]   = {(const float*)d_in[5],  (const float*)d_in[13], (const float*)d_in[21]};
    const float* w[3]     = {(const float*)d_in[6],  (const float*)d_in[14], (const float*)d_in[22]};
    const float* disc[3]  = {(const float*)d_in[7],  (const float*)d_in[15], (const float*)d_in[23]};

    // Workspace layout: 768 floats of ||w||^2, then bf16 hi/lo planes
    // (w_hi, w_lo, beta_hi, beta_lo) per source.  Total ~1.75 MB + 3 KB.
    const int P[3] = {512, 256, 128};
    float*  wn     = (float*)d_ws;
    __bf16* bfb    = (__bf16*)((char*)d_ws + 768 * sizeof(float));
    __bf16 *wh[3], *wl[3], *bh[3], *bl[3];
    size_t  o = 0;
    for (int s = 0; s < 3; ++s) {
        size_t n = (size_t)KDIM * P[s];
        wh[s] = bfb + o; o += n;
        wl[s] = bfb + o; o += n;
        bh[s] = bfb + o; o += n;
        bl[s] = bfb + o; o += n;
    }

    for (int s = 0; s < 3; ++s)
        split_kernel<<<KDIM, 128, 0, stream>>>(w[s], beta[s], wh[s], wl[s],
                                               bh[s], bl[s], wn + s * KDIM, P[s]);

    source_kernel<512><<<NT / 32, 256, 0, stream>>>(
        x[0], alpha[0], sig[0], eta[0], gam[0],
        wh[0], wl[0], bh[0], bl[0], wn + 0 * KDIM, out, 0);
    source_kernel<256><<<NT / 32, 256, 0, stream>>>(
        x[1], alpha[1], sig[1], eta[1], gam[1],
        wh[1], wl[1], bh[1], bl[1], wn + 1 * KDIM, out, 1);
    source_kernel<128><<<NT / 32, 256, 0, stream>>>(
        x[2], alpha[2], sig[2], eta[2], gam[2],
        wh[2], wl[2], bh[2], bl[2], wn + 2 * KDIM, out, 2);

    combine_kernel<<<NT / 256, 256, 0, stream>>>(out, disc[0], disc[1], disc[2]);
}
